// iLoRAModel_2869038154552
// MI455X (gfx1250) — matrix operand order repaired and checked by reading, not compiled, not run on hardware
//
#include <hip/hip_runtime.h>
#include <hip/hip_bf16.h>

// ---------------------------------------------------------------------------
// Problem constants (from the reference)
// ---------------------------------------------------------------------------
#define B_        16
#define L_        2048
#define D_        2048
#define S_        50
#define HREC_     64
#define PROJ_HID_ 2048
#define GATE_HID_ 128
#define NEXP_     8
#define HIS_ID    32000

typedef __attribute__((ext_vector_type(16))) __bf16 v16bf;
typedef __attribute__((ext_vector_type(8)))  float  v8f;
typedef __attribute__((ext_vector_type(4)))  float  f4;

// ---------------------------------------------------------------------------
// Kernel 1: gate MLP + softmax.  [16,64] -> relu([16,128]) -> [16,8] -> softmax
// ---------------------------------------------------------------------------
__global__ __launch_bounds__(GATE_HID_) void gate_kernel(
    const float* __restrict__ last, const float* __restrict__ Wg1,
    const float* __restrict__ bg1,  const float* __restrict__ Wg2,
    const float* __restrict__ bg2,  float* __restrict__ gate_out) {
  const int b = blockIdx.x;
  const int t = threadIdx.x;            // 0..127
  __shared__ float h[GATE_HID_];
  __shared__ float logits[NEXP_];
  const float* x = last + b * HREC_;
  float acc = bg1[t];
  #pragma unroll 8
  for (int k = 0; k < HREC_; ++k) acc += x[k] * Wg1[k * GATE_HID_ + t];
  h[t] = fmaxf(acc, 0.0f);
  __syncthreads();
  if (t < NEXP_) {
    float a = bg2[t];
    for (int k = 0; k < GATE_HID_; ++k) a += h[k] * Wg2[k * NEXP_ + t];
    logits[t] = a;
  }
  __syncthreads();
  if (t == 0) {
    float mx = logits[0];
    for (int e = 1; e < NEXP_; ++e) mx = fmaxf(mx, logits[e]);
    float ex[NEXP_], s = 0.0f;
    for (int e = 0; e < NEXP_; ++e) { ex[e] = __expf(logits[e] - mx); s += ex[e]; }
    const float inv = 1.0f / s;
    for (int e = 0; e < NEXP_; ++e) gate_out[b * NEXP_ + e] = ex[e] * inv;
  }
}

// ---------------------------------------------------------------------------
// Fragment packing: fp32 -> bf16 hi/lo pairs, laid out in the exact WMMA
// per-lane register order (ISA 7.12.2), fragment-major:
//   frag index = (tileRow * tilesInner + tileInner) * 32 + lane, 16 bf16 each.
// GEMM lanes then load one contiguous 32-byte chunk per fragment operand.
// ---------------------------------------------------------------------------

// A matrix 16x32 bf16 layout: lane half 0 -> K {0..7,16..23}, half 1 ->
// K {8..15,24..31}; VGPR v holds K pair (2v,2v+1).
__global__ __launch_bounds__(32) void pack_a_frags(
    const float* __restrict__ X, int K,
    __bf16* __restrict__ hiOut, __bf16* __restrict__ loOut) {
  const int kt = blockIdx.x;            // K tile (32 wide)
  const int mt = blockIdx.y;            // M tile (16 tall)
  const int Kt = K / 32;
  const int lane = threadIdx.x;
  const int half = lane >> 4, lid = lane & 15;
  const int row = mt * 16 + lid;
  v16bf hi, lo;
  #pragma unroll
  for (int e = 0; e < 16; ++e) {
    const int v = e >> 1, lob = e & 1;
    const int koff = ((v >> 2) * 16) + half * 8 + ((v & 3) * 2) + lob;
    const float f = X[(size_t)row * K + kt * 32 + koff];
    const __bf16 h = (__bf16)f;
    hi[e] = h;
    lo[e] = (__bf16)(f - (float)h);
  }
  const size_t idx = (size_t)(mt * Kt + kt) * 32 + lane;
  ((v16bf*)hiOut)[idx] = hi;
  ((v16bf*)loOut)[idx] = lo;
}

// B matrix 32x16 bf16 layout: lanes 0-15 hold K {0..7,16..23} (col = lane&15),
// lanes 16-31 hold K {8..15,24..31}; same K order as A.
__global__ __launch_bounds__(32) void pack_b_frags(
    const float* __restrict__ W, int K, int N,
    __bf16* __restrict__ hiOut, __bf16* __restrict__ loOut) {
  const int nt = blockIdx.x;            // N tile (16 wide)
  const int kt = blockIdx.y;            // K tile (32 tall)
  const int Ntiles = N / 16;
  const int lane = threadIdx.x;
  const int half = lane >> 4, col = nt * 16 + (lane & 15);
  v16bf hi, lo;
  #pragma unroll
  for (int e = 0; e < 16; ++e) {
    const float f = W[(size_t)(kt * 32 + (e >> 3) * 16 + half * 8 + (e & 7)) * N + col];
    const __bf16 h = (__bf16)f;
    hi[e] = h;
    lo[e] = (__bf16)(f - (float)h);
  }
  const size_t idx = (size_t)(kt * Ntiles + nt) * 32 + lane;
  ((v16bf*)hiOut)[idx] = hi;
  ((v16bf*)loOut)[idx] = lo;
}

// ---------------------------------------------------------------------------
// WMMA GEMM on pre-packed fragments:  Y = act(A @ W + bias)
// Split-precision bf16: acc += Ahi*Bhi + Ahi*Blo + Alo*Bhi (f32 accumulate)
// Inner loop: 4 contiguous 32B lane loads + 3 v_wmma_f32_16x16x32_bf16.
// One wave32 per 16x16 output tile.
// ---------------------------------------------------------------------------
__global__ __launch_bounds__(32) void wmma_gemm_packed(
    const __bf16* __restrict__ Ahi, const __bf16* __restrict__ Alo,
    const __bf16* __restrict__ Bhi, const __bf16* __restrict__ Blo,
    const float* __restrict__ bias, float* __restrict__ Y,
    int N, int K, int doRelu) {
  const int nt = blockIdx.x;
  const int mt = blockIdx.y;
  const int lane = threadIdx.x & 31;
  const int Kt = K / 32, Ntiles = N / 16;
  const v16bf* pAhi = (const v16bf*)Ahi;
  const v16bf* pAlo = (const v16bf*)Alo;
  const v16bf* pBhi = (const v16bf*)Bhi;
  const v16bf* pBlo = (const v16bf*)Blo;

  size_t aIdx = (size_t)mt * Kt * 32 + lane;
  size_t bIdx = (size_t)nt * 32 + lane;
  const size_t bStep = (size_t)Ntiles * 32;

  v8f acc = {};
  for (int kt = 0; kt < Kt; ++kt) {
    const v16bf ah = pAhi[aIdx];
    const v16bf al = pAlo[aIdx];
    const v16bf bh = pBhi[bIdx];
    const v16bf bl = pBlo[bIdx];
    acc = __builtin_amdgcn_wmma_f32_16x16x32_bf16(false, ah, false, bh,
                                                  (short)0, acc, false, false);
    acc = __builtin_amdgcn_wmma_f32_16x16x32_bf16(false, ah, false, bl,
                                                  (short)0, acc, false, false);
    acc = __builtin_amdgcn_wmma_f32_16x16x32_bf16(false, al, false, bh,
                                                  (short)0, acc, false, false);
    aIdx += 32;
    bIdx += bStep;
  }

  // C/D layout: VGPR i -> M = i (lanes 0-15) / 8+i (lanes 16-31), N = col.
  const int half = lane >> 4, lid = lane & 15;
  const int col = nt * 16 + lid;
  const float bv = bias[col];
  #pragma unroll
  for (int i = 0; i < 8; ++i) {
    float v = acc[i] + bv;
    if (doRelu) v = fmaxf(v, 0.0f);
    Y[(size_t)(mt * 16 + half * 8 + i) * N + col] = v;
  }
}

// ---------------------------------------------------------------------------
// Kernel 3a: per-batch rank matching.  srcIdx[b,l] = -1 (keep input embed) or
// s index of the item-history row replacing position l.
// ---------------------------------------------------------------------------
__global__ void build_src_idx(const int* __restrict__ input_ids,
                              const int* __restrict__ item_seq,
                              const int* __restrict__ len_seq,
                              int* __restrict__ srcIdx) {
  const int b = blockIdx.x;
  if (threadIdx.x != 0) return;
  const int* ids   = input_ids + b * L_;
  const int* items = item_seq + b * S_;
  const int  len   = len_seq[b];

  int vlist[S_ + 1];                    // vlist[k] = s of k-th-from-end valid
  int nv = 0;
  for (int s = S_ - 1; s >= 0; --s) {
    const bool valid = (s >= S_ - len) && (items[s] != 0);
    if (valid) { ++nv; vlist[nv] = s; }
  }
  int np = 0;
  for (int l = 0; l < L_; ++l) np += (ids[l] == HIS_ID) ? 1 : 0;
  const int r = nv < np ? nv : np;

  int* dst = srcIdx + b * L_;
  int k = 0;
  for (int l = L_ - 1; l >= 0; --l) {
    int src = -1;
    if (ids[l] == HIS_ID) { ++k; if (k <= r) src = vlist[k]; }
    dst[l] = src;
  }
}

// ---------------------------------------------------------------------------
// Kernel 3b: bandwidth phase.  One block per (b,l) row of 2048 floats.
// 537 MB traffic -> ~23 us at 23.3 TB/s.  Non-temporal: 268 MB output > 192 MB L2.
// ---------------------------------------------------------------------------
__global__ __launch_bounds__(256) void scatter_copy(
    const float* __restrict__ input_embeds, const float* __restrict__ his,
    const int* __restrict__ srcIdx, float* __restrict__ out) {
  const size_t row = blockIdx.x;        // b*L + l
  const int b = (int)(row / L_);
  const int src = srcIdx[row];
  const f4* in = (src < 0)
      ? (const f4*)(input_embeds + row * D_)
      : (const f4*)(his + (size_t)(b * S_ + src) * D_);
  f4* o = (f4*)(out + row * D_);
  const int t = threadIdx.x;            // 512 float4 per row, 2 per thread
  const f4 v0 = __builtin_nontemporal_load(&in[t]);
  const f4 v1 = __builtin_nontemporal_load(&in[t + 256]);
  __builtin_nontemporal_store(v0, &o[t]);
  __builtin_nontemporal_store(v1, &o[t + 256]);
}

// ---------------------------------------------------------------------------
// Launch
// ---------------------------------------------------------------------------
extern "C" void kernel_launch(void* const* d_in, const int* in_sizes, int n_in,
                              void* d_out, int out_size, void* d_ws, size_t ws_size,
                              hipStream_t stream) {
  (void)in_sizes; (void)n_in; (void)out_size; (void)ws_size;
  const int*   input_ids     = (const int*)d_in[0];
  const int*   item_seq      = (const int*)d_in[1];
  const int*   len_seq       = (const int*)d_in[2];
  const float* input_embeds  = (const float*)d_in[3];
  const float* full_seq_rec  = (const float*)d_in[4];   // [800, 64]
  const float* last_item_rep = (const float*)d_in[5];
  const float* Wp1 = (const float*)d_in[6];
  const float* bp1 = (const float*)d_in[7];
  const float* Wp2 = (const float*)d_in[8];
  const float* bp2 = (const float*)d_in[9];
  const float* Wg1 = (const float*)d_in[10];
  const float* bg1 = (const float*)d_in[11];
  const float* Wg2 = (const float*)d_in[12];
  const float* bg2 = (const float*)d_in[13];

  float* out_embeds = (float*)d_out;                           // [B,L,D]
  float* gate_out   = out_embeds + (size_t)B_ * L_ * D_;       // [B,NEXP]

  const int M   = B_ * S_;        // 800
  const int Mt  = M / 16;         // 50
  const int Kt1 = HREC_ / 32;     // 2
  const int Nt1 = PROJ_HID_ / 16; // 128
  const int Kt2 = PROJ_HID_ / 32; // 64
  const int Nt2 = D_ / 16;        // 128

  // ---- workspace carve-up (all region sizes are multiples of 256 bytes) ----
  char* ws = (char*)d_ws;
  float* hidden = (float*)ws;  ws += (size_t)M * PROJ_HID_ * 4;        // 6.55 MB
  float* his    = (float*)ws;  ws += (size_t)M * D_ * 4;               // 6.55 MB
  int* srcIdx   = (int*)ws;    ws += (size_t)B_ * L_ * 4;              // 131 KB
  const size_t fragA1 = (size_t)Mt  * Kt1 * 32 * 16;  // elements
  const size_t fragA2 = (size_t)Mt  * Kt2 * 32 * 16;
  const size_t fragB1 = (size_t)Kt1 * Nt1 * 32 * 16;
  const size_t fragB2 = (size_t)Kt2 * Nt2 * 32 * 16;
  __bf16* pA1hi = (__bf16*)ws; ws += fragA1 * 2;
  __bf16* pA1lo = (__bf16*)ws; ws += fragA1 * 2;
  __bf16* pA2hi = (__bf16*)ws; ws += fragA2 * 2;
  __bf16* pA2lo = (__bf16*)ws; ws += fragA2 * 2;
  __bf16* pB1hi = (__bf16*)ws; ws += fragB1 * 2;
  __bf16* pB1lo = (__bf16*)ws; ws += fragB1 * 2;
  __bf16* pB2hi = (__bf16*)ws; ws += fragB2 * 2;
  __bf16* pB2lo = (__bf16*)ws; ws += fragB2 * 2;

  // 1) gate MLP + softmax -> tail of d_out
  gate_kernel<<<B_, GATE_HID_, 0, stream>>>(last_item_rep, Wg1, bg1, Wg2, bg2,
                                            gate_out);

  // 2) pack weights + layer-1 activations into WMMA fragment layout
  pack_b_frags<<<dim3(Nt1, Kt1), 32, 0, stream>>>(Wp1, HREC_, PROJ_HID_,
                                                  pB1hi, pB1lo);
  pack_b_frags<<<dim3(Nt2, Kt2), 32, 0, stream>>>(Wp2, PROJ_HID_, D_,
                                                  pB2hi, pB2lo);
  pack_a_frags<<<dim3(Kt1, Mt), 32, 0, stream>>>(full_seq_rec, HREC_,
                                                 pA1hi, pA1lo);

  // 3) layer 1: hidden = relu(full_seq_rec @ Wp1 + bp1)
  wmma_gemm_packed<<<dim3(Nt1, Mt), 32, 0, stream>>>(
      pA1hi, pA1lo, pB1hi, pB1lo, bp1, hidden, PROJ_HID_, HREC_, /*relu=*/1);

  // 4) repack hidden, layer 2: his = hidden @ Wp2 + bp2
  pack_a_frags<<<dim3(Kt2, Mt), 32, 0, stream>>>(hidden, PROJ_HID_,
                                                 pA2hi, pA2lo);
  wmma_gemm_packed<<<dim3(Nt2, Mt), 32, 0, stream>>>(
      pA2hi, pA2lo, pB2hi, pB2lo, bp2, his, D_, PROJ_HID_, /*relu=*/0);

  // 5) rank matching + bandwidth-bound scatter copy
  build_src_idx<<<B_, 32, 0, stream>>>(input_ids, item_seq, len_seq, srcIdx);
  scatter_copy<<<B_ * L_, 256, 0, stream>>>(input_embeds, his, srcIdx, out_embeds);
}
